// MultiHeadAttention_40656160424152
// MI455X (gfx1250) — matrix operand order repaired and checked by reading, not compiled, not run on hardware
//
#include <hip/hip_runtime.h>
#include <hip/hip_bf16.h>

typedef __attribute__((ext_vector_type(16))) _Float16 v16h;
typedef __attribute__((ext_vector_type(8)))  _Float16 v8h;
typedef __attribute__((ext_vector_type(4)))  _Float16 v4h;
typedef __attribute__((ext_vector_type(8)))  float    v8f;

#define D_MODEL 1024
#define SEQ     1024
#define NBATCH  4
#define NHEAD   16
#define DEPTH   64

// ---------------------------------------------------------------------------
// WMMA helpers (V_WMMA_F32_16X16X32_F16, wave32)
// ---------------------------------------------------------------------------
__device__ __forceinline__ v8f wmma16x16x32(v16h a, v16h b, v8f c) {
    return __builtin_amdgcn_wmma_f32_16x16x32_f16(
        /*neg_a=*/false, a, /*neg_b=*/false, b,
        /*c_mod=*/(short)0, c, /*reuse_a=*/false, /*reuse_b=*/false);
}

__device__ __forceinline__ v8f zero8() {
    v8f z;
#pragma unroll
    for (int i = 0; i < 8; ++i) z[i] = 0.0f;
    return z;
}

// ---------------------------------------------------------------------------
// Async global->LDS staging (GLOBAL_LOAD_ASYNC_TO_LDS_B128, ASYNCcnt).
// LDS operand = low 32 bits of the generic pointer (LDS offset space).
// ---------------------------------------------------------------------------
__device__ __forceinline__ unsigned lds_off_u32(const void* p) {
    return (unsigned)(unsigned long long)p;
}
__device__ __forceinline__ void async_ld_b128(unsigned ldsOff, const void* g) {
    asm volatile("global_load_async_to_lds_b128 %0, %1, off"
                 :: "v"(ldsOff), "v"(g) : "memory");
}
__device__ __forceinline__ void wait_async0() {
    asm volatile("s_wait_asynccnt 0" ::: "memory");
}

// A-fragment: 16x32 f16 tile, row-major in LDS with row stride `ld` halves.
// ISA layout: lanes 0-15 (M=lane):  elems 0-7 -> K 0-7,  elems 8-15 -> K 16-23
//             lanes 16-31 (M=l-16): elems 0-7 -> K 8-15, elems 8-15 -> K 24-31
__device__ __forceinline__ v16h load_frag_a(const _Float16* p, int ld) {
    const int lane = threadIdx.x & 31;
    const int m    = lane & 15;
    const int kb   = (lane >> 4) << 3;          // 0 or 8
    const _Float16* q = p + m * ld + kb;
    v8h lo = *(const v8h*)(q);
    v8h hi = *(const v8h*)(q + 16);
    v16h r;
#pragma unroll
    for (int i = 0; i < 8; ++i) { r[i] = lo[i]; r[i + 8] = hi[i]; }
    return r;
}

// B-fragment: operand column n = lane&15; lanes 0-15: elems 0-7 -> K 0-7, elems 8-15 -> K 16-23,
// lanes 16-31: elems 0-7 -> K 8-15, elems 8-15 -> K 24-31. Source tile is row-major
// "B-transposed": row = n, cols = K, row stride `ld` halves.
__device__ __forceinline__ v16h load_frag_b(const _Float16* p, int ld) {
    const int lane = threadIdx.x & 31;
    const int n    = lane & 15;
    const int kb   = (lane >> 4) << 3;          // 0 or 8
    const _Float16* q = p + n * ld + kb;
    v8h lo = *(const v8h*)(q);
    v8h hi = *(const v8h*)(q + 16);
    v16h r;
#pragma unroll
    for (int i = 0; i < 8; ++i) { r[i] = lo[i]; r[i + 8] = hi[i]; }
    return r;
}

// ---------------------------------------------------------------------------
// Kernel 1: weight transpose + f32 -> f16 convert.  Wt[n][k] = (f16)W[k][n]
// ---------------------------------------------------------------------------
__global__ __launch_bounds__(256) void wtrans_kernel(const float* __restrict__ W,
                                                     _Float16* __restrict__ Wt) {
    __shared__ float t[32][33];
    const int tx = threadIdx.x & 31;
    const int ty = threadIdx.x >> 5;            // 0..7
    const int n0 = blockIdx.x * 32;
    const int k0 = blockIdx.y * 32;
#pragma unroll
    for (int i = 0; i < 32; i += 8)
        t[ty + i][tx] = W[(size_t)(k0 + ty + i) * D_MODEL + n0 + tx];
    __syncthreads();
#pragma unroll
    for (int i = 0; i < 32; i += 8)
        Wt[(size_t)(n0 + ty + i) * D_MODEL + k0 + tx] = (_Float16)t[tx][ty + i];
}

// ---------------------------------------------------------------------------
// Kernel 2: tiled WMMA GEMM  Y[4096 x 1024] = A[4096 x 1024] @ Wt^T + bias
//   AMODE 0: A is f32 row-major (convert to f16 through LDS, VALU path)
//   AMODE 1: A is f16 row-major (async global->LDS)
//   OMODE 0: write f16 head-split   out[(b*16+h)*1024 + s][d]      (Q, K)
//   OMODE 1: write f16 head-split-T out[(b*16+h)*64 + d][s]        (V)
//   OMODE 2: write f32 row-major    out[m][n]                      (final)
// Block: 256 threads (8 waves), 128x128 tile; wave = 32x64 (2x4 WMMA tiles).
// ---------------------------------------------------------------------------
template <int AMODE, int OMODE>
__global__ __launch_bounds__(256) void gemm_kernel(const void* __restrict__ Aptr,
                                                   const _Float16* __restrict__ Wt,
                                                   const float* __restrict__ bias,
                                                   void* __restrict__ outp) {
    constexpr int LDA = 48;                     // 32 + 16 pad, 96B rows (16B mult)
    __shared__ _Float16 sA[128 * LDA];
    __shared__ _Float16 sB[128 * LDA];

    const int tid  = threadIdx.x;
    const int lane = tid & 31;
    const int wave = tid >> 5;                  // 0..7
    const int wm   = wave & 3;                  // 4 row groups of 32
    const int wn   = wave >> 2;                 // 2 col groups of 64
    const int m0   = blockIdx.y * 128;
    const int n0   = blockIdx.x * 128;

    v8f acc[2][4];
#pragma unroll
    for (int tm = 0; tm < 2; ++tm)
#pragma unroll
        for (int tn = 0; tn < 4; ++tn) acc[tm][tn] = zero8();

    for (int k0 = 0; k0 < D_MODEL; k0 += 32) {
        __syncthreads();
        if (AMODE == 0) {
            const float* A = (const float*)Aptr;
            for (int i = tid; i < 128 * 8; i += 256) {
                const int r = i >> 3, c = (i & 7) * 4;
                const float4 v = *(const float4*)&A[(size_t)(m0 + r) * D_MODEL + k0 + c];
                v4h h; h[0] = (_Float16)v.x; h[1] = (_Float16)v.y;
                       h[2] = (_Float16)v.z; h[3] = (_Float16)v.w;
                *(v4h*)&sA[r * LDA + c] = h;
            }
        } else {
            const _Float16* A = (const _Float16*)Aptr;
            for (int i = tid; i < 128 * 4; i += 256) {
                const int r = i >> 2, c = (i & 3) * 8;
                async_ld_b128(lds_off_u32(&sA[r * LDA + c]),
                              &A[(size_t)(m0 + r) * D_MODEL + k0 + c]);
            }
        }
        for (int i = tid; i < 128 * 4; i += 256) {
            const int r = i >> 2, c = (i & 3) * 8;
            async_ld_b128(lds_off_u32(&sB[r * LDA + c]),
                          &Wt[(size_t)(n0 + r) * D_MODEL + k0 + c]);
        }
        wait_async0();
        __syncthreads();

        v16h af[2], bf[4];
#pragma unroll
        for (int tm = 0; tm < 2; ++tm)
            af[tm] = load_frag_a(&sA[(wm * 32 + tm * 16) * LDA], LDA);
#pragma unroll
        for (int tn = 0; tn < 4; ++tn)
            bf[tn] = load_frag_b(&sB[(wn * 64 + tn * 16) * LDA], LDA);
#pragma unroll
        for (int tm = 0; tm < 2; ++tm)
#pragma unroll
            for (int tn = 0; tn < 4; ++tn)
                acc[tm][tn] = wmma16x16x32(af[tm], bf[tn], acc[tm][tn]);
    }

    // Epilogue. C/D layout: vgpr j -> M = j (+8 for upper half-wave), N = lane&15.
    const int nl = lane & 15;
    const int mh = (lane & 16) ? 8 : 0;
#pragma unroll
    for (int tm = 0; tm < 2; ++tm)
#pragma unroll
        for (int tn = 0; tn < 4; ++tn)
#pragma unroll
            for (int j = 0; j < 8; ++j) {
                const int m = m0 + wm * 32 + tm * 16 + mh + j;
                const int n = n0 + wn * 64 + tn * 16 + nl;
                const float v = acc[tm][tn][j] + bias[n];
                if (OMODE == 2) {
                    ((float*)outp)[(size_t)m * D_MODEL + n] = v;
                } else {
                    const int b = m >> 10, s = m & 1023;
                    const int h = n >> 6,  d = n & 63;
                    _Float16* o = (_Float16*)outp;
                    if (OMODE == 0)
                        o[((size_t)(b * NHEAD + h) * SEQ + s) * DEPTH + d] = (_Float16)v;
                    else
                        o[((size_t)(b * NHEAD + h) * DEPTH + d) * SEQ + s] = (_Float16)v;
                }
            }
}

// ---------------------------------------------------------------------------
// Kernel 3: causal flash attention, one (128-query block, b*h) per workgroup.
// 8 waves; each wave owns 16 query rows. Streams 64-key K/V tiles via
// async global->LDS copies.
// ---------------------------------------------------------------------------
__global__ __launch_bounds__(256) void attn_kernel(const _Float16* __restrict__ Qh,
                                                   const _Float16* __restrict__ Kh,
                                                   const _Float16* __restrict__ Vt,
                                                   _Float16* __restrict__ ctx) {
    constexpr int LD = 72;                      // 64 + 8 pad, 144B rows (16B mult)
    __shared__ _Float16 sQ[128 * LD];
    __shared__ _Float16 sK[64 * LD];
    __shared__ _Float16 sV[64 * LD];
    __shared__ _Float16 sP[128 * LD];           // per-wave 16-row P scratch

    const int tid  = threadIdx.x;
    const int lane = tid & 31;
    const int wave = tid >> 5;
    const int qb   = blockIdx.x;                // 0..7  (query block of 128)
    const int bh   = blockIdx.y;                // 0..63 (b*16 + h)
    const size_t qkbase = (size_t)bh * SEQ * DEPTH;
    const size_t vbase  = (size_t)bh * DEPTH * SEQ;

    // Resident Q tile (128 x 64 halves), async; completion covered by the
    // first in-loop s_wait_asynccnt 0.
    for (int i = tid; i < 128 * 8; i += 256) {
        const int r = i >> 3, c = (i & 7) * 8;
        async_ld_b128(lds_off_u32(&sQ[r * LD + c]),
                      &Qh[qkbase + (size_t)(qb * 128 + r) * DEPTH + c]);
    }

    v8f oacc[4];
    float mstate[8], lstate[8];
#pragma unroll
    for (int tn = 0; tn < 4; ++tn) oacc[tn] = zero8();
#pragma unroll
    for (int j = 0; j < 8; ++j) { mstate[j] = -1e30f; lstate[j] = 0.0f; }

    const int nl = lane & 15;
    const int mh = (lane & 16) ? 8 : 0;
    const int qrow0 = qb * 128 + wave * 16 + mh;

    const int nkb = qb * 2 + 2;                 // causal: only up to diagonal
    for (int kb = 0; kb < nkb; ++kb) {
        __syncthreads();                        // protect sK/sV reuse
        for (int i = tid; i < 64 * 8; i += 256) {
            const int r = i >> 3, c = (i & 7) * 8;
            async_ld_b128(lds_off_u32(&sK[r * LD + c]),
                          &Kh[qkbase + (size_t)(kb * 64 + r) * DEPTH + c]);
            async_ld_b128(lds_off_u32(&sV[r * LD + c]),
                          &Vt[vbase + (size_t)r * SEQ + kb * 64 + c]);
        }
        wait_async0();
        __syncthreads();

        // S = Q @ K^T  (16 x 64 per wave)
        v8f sacc[4];
#pragma unroll
        for (int tn = 0; tn < 4; ++tn) sacc[tn] = zero8();
#pragma unroll
        for (int ks = 0; ks < 2; ++ks) {
            const v16h aq = load_frag_a(&sQ[(wave * 16) * LD + ks * 32], LD);
#pragma unroll
            for (int tn = 0; tn < 4; ++tn) {
                const v16h bk = load_frag_b(&sK[(tn * 16) * LD + ks * 32], LD);
                sacc[tn] = wmma16x16x32(aq, bk, sacc[tn]);
            }
        }

        // scale, causal mask, online softmax
        const int kcol0 = kb * 64;
#pragma unroll
        for (int j = 0; j < 8; ++j) {
            float sv[4];
#pragma unroll
            for (int tn = 0; tn < 4; ++tn) {
                float s = sacc[tn][j] * 0.125f;             // 1/sqrt(64)
                if (kcol0 + tn * 16 + nl > qrow0 + j) s -= 1e9f;
                sv[tn] = s;
            }
            float rmax = fmaxf(fmaxf(sv[0], sv[1]), fmaxf(sv[2], sv[3]));
            rmax = fmaxf(rmax, __shfl_xor(rmax, 1, 32));
            rmax = fmaxf(rmax, __shfl_xor(rmax, 2, 32));
            rmax = fmaxf(rmax, __shfl_xor(rmax, 4, 32));
            rmax = fmaxf(rmax, __shfl_xor(rmax, 8, 32));
            const float newm = fmaxf(mstate[j], rmax);
            const float fac  = __expf(mstate[j] - newm);
            mstate[j] = newm;
            float rsum = 0.0f;
#pragma unroll
            for (int tn = 0; tn < 4; ++tn) {
                const float p = __expf(sv[tn] - newm);
                rsum += p;
                sP[(wave * 16 + mh + j) * LD + tn * 16 + nl] = (_Float16)p;
            }
            rsum += __shfl_xor(rsum, 1, 32);
            rsum += __shfl_xor(rsum, 2, 32);
            rsum += __shfl_xor(rsum, 4, 32);
            rsum += __shfl_xor(rsum, 8, 32);
            lstate[j] = lstate[j] * fac + rsum;
#pragma unroll
            for (int tn = 0; tn < 4; ++tn) oacc[tn][j] *= fac;
        }
        // wave-private P region; same-wave LDS ops are in-order -> no barrier

        // O += P @ V   (V^T rows are d, cols are key -> B-fragments)
#pragma unroll
        for (int ks = 0; ks < 2; ++ks) {
            const v16h ap = load_frag_a(&sP[(wave * 16) * LD + ks * 32], LD);
#pragma unroll
            for (int tn = 0; tn < 4; ++tn) {
                const v16h bv = load_frag_b(&sV[(tn * 16) * LD + ks * 32], LD);
                oacc[tn] = wmma16x16x32(ap, bv, oacc[tn]);
            }
        }
    }

    // normalize and write merged-head ctx (f16, [B,S,D])
    const int b = bh >> 4, h = bh & 15;
#pragma unroll
    for (int tn = 0; tn < 4; ++tn)
#pragma unroll
        for (int j = 0; j < 8; ++j) {
            const int q = qrow0 + j;
            const int d = tn * 16 + nl;
            const float v = oacc[tn][j] / lstate[j];
            ctx[((size_t)b * SEQ + q) * D_MODEL + h * DEPTH + d] = (_Float16)v;
        }
}

// ---------------------------------------------------------------------------
// Host launcher
// ---------------------------------------------------------------------------
extern "C" void kernel_launch(void* const* d_in, const int* in_sizes, int n_in,
                              void* d_out, int out_size, void* d_ws, size_t ws_size,
                              hipStream_t stream) {
    (void)in_sizes; (void)n_in; (void)out_size; (void)ws_size;
    const float* q    = (const float*)d_in[0];
    const float* k    = (const float*)d_in[1];
    const float* v    = (const float*)d_in[2];
    /* d_in[3] = mask: causality handled analytically */
    const float* wq   = (const float*)d_in[4];
    const float* bq   = (const float*)d_in[5];
    const float* wk   = (const float*)d_in[6];
    const float* bk   = (const float*)d_in[7];
    const float* wv   = (const float*)d_in[8];
    const float* bv   = (const float*)d_in[9];
    const float* wo   = (const float*)d_in[10];
    const float* bo   = (const float*)d_in[11];
    float* out = (float*)d_out;

    const size_t MB = 1024ull * 1024ull;
    char* ws = (char*)d_ws;
    _Float16* wqT = (_Float16*)(ws + 0 * MB);   // 1M halves each
    _Float16* wkT = (_Float16*)(ws + 2 * MB);
    _Float16* wvT = (_Float16*)(ws + 4 * MB);
    _Float16* woT = (_Float16*)(ws + 6 * MB);
    _Float16* Qh  = (_Float16*)(ws + 8 * MB);   // [B,H,S,64]  8MB
    _Float16* Kh  = (_Float16*)(ws + 16 * MB);  // [B,H,S,64]  8MB
    _Float16* Vt  = (_Float16*)(ws + 24 * MB);  // [B,H,64,S]  8MB
    _Float16* ctx = (_Float16*)(ws + 32 * MB);  // [B,S,D]     8MB

    const dim3 blk(256);
    // 1) weight transpose/convert
    wtrans_kernel<<<dim3(32, 32), blk, 0, stream>>>(wq, wqT);
    wtrans_kernel<<<dim3(32, 32), blk, 0, stream>>>(wk, wkT);
    wtrans_kernel<<<dim3(32, 32), blk, 0, stream>>>(wv, wvT);
    wtrans_kernel<<<dim3(32, 32), blk, 0, stream>>>(wo, woT);

    // 2) QKV projections (f32 A -> WMMA -> f16 head-split outputs)
    const dim3 ggrid(D_MODEL / 128, (NBATCH * SEQ) / 128);   // 8 x 32
    gemm_kernel<0, 0><<<ggrid, blk, 0, stream>>>(q, wqT, bq, Qh);
    gemm_kernel<0, 0><<<ggrid, blk, 0, stream>>>(k, wkT, bk, Kh);
    gemm_kernel<0, 1><<<ggrid, blk, 0, stream>>>(v, wvT, bv, Vt);

    // 3) causal flash attention
    attn_kernel<<<dim3(SEQ / 128, NBATCH * NHEAD), blk, 0, stream>>>(Qh, Kh, Vt, ctx);

    // 4) output projection (f16 A -> f32 out + bias)
    gemm_kernel<1, 2><<<ggrid, blk, 0, stream>>>(ctx, woT, bo, out);
}
